// PointPatchEmbed_1468878815319
// MI455X (gfx1250) — matrix operand order repaired and checked by reading, not compiled, not run on hardware
//
#include <hip/hip_runtime.h>

// ---- problem constants (match reference) ----
#define BB     8
#define NN     8192
#define MCENT  512
#define KNB    32
#define LCOLS  (BB*MCENT*KNB)   // 131072 = 2^17
#define R2     0.01f
#define EPSV   1e-5f

typedef __attribute__((ext_vector_type(16))) __bf16 v16bf;
typedef __attribute__((ext_vector_type(8)))  __bf16 v8bf;
typedef __attribute__((ext_vector_type(8)))  float  v8f;
typedef __attribute__((ext_vector_type(4)))  int    v4i;

#if defined(__gfx1250__) && __has_builtin(__builtin_amdgcn_global_load_async_to_lds_b128) && __has_builtin(__builtin_amdgcn_s_wait_asynccnt)
#define USE_ASYNC 1
typedef __attribute__((address_space(1))) v4i* gv4i_p;   // global int4*
typedef __attribute__((address_space(3))) v4i* sv4i_p;   // LDS int4*
#else
#define USE_ASYNC 0
#endif

// ---------------- pass-through copy ----------------
__global__ __launch_bounds__(256) void copy_f32(const float* __restrict__ src,
                                                float* __restrict__ dst, int n) {
  int i = blockIdx.x * 256 + threadIdx.x;
  if (i < n) dst[i] = src[i];
}

// ---------------- pad W0 [256][6] -> [256][32] ----------------
__global__ __launch_bounds__(256) void pad_w0(const float* __restrict__ W0,
                                              float* __restrict__ Wpad) {
  int i = blockIdx.x * 256 + threadIdx.x;     // 256*32 = 8192
  int r = i >> 5, c = i & 31;
  Wpad[i] = (c < 6) ? W0[r * 6 + c] : 0.0f;
}

// ---------------- furthest point sampling: one block per batch ----------------
__global__ __launch_bounds__(256) void fps_kernel(const float* __restrict__ p,
                                                  int* __restrict__ idx_out,
                                                  float* __restrict__ centers) {
  __shared__ float min_d[NN];
  __shared__ float rv[256];
  __shared__ int   ri[256];
  int b = blockIdx.x, tid = threadIdx.x;
  const float* pb = p + (size_t)b * NN * 3;
  for (int n = tid; n < NN; n += 256) min_d[n] = 1e10f;
  __syncthreads();
  int last = 0;
  for (int j = 0; j < MCENT; ++j) {
    float lx = pb[last*3+0], ly = pb[last*3+1], lz = pb[last*3+2];
    if (tid == 0) {
      idx_out[b*MCENT + j] = last;
      centers[(b*MCENT + j)*3 + 0] = lx;
      centers[(b*MCENT + j)*3 + 1] = ly;
      centers[(b*MCENT + j)*3 + 2] = lz;
    }
    float best = -1.0f; int bi = 0;
    for (int n = tid; n < NN; n += 256) {
      float dx = pb[n*3+0]-lx, dy = pb[n*3+1]-ly, dz = pb[n*3+2]-lz;
      float d  = dx*dx + dy*dy + dz*dz;
      float md = fminf(min_d[n], d);
      min_d[n] = md;
      if (md > best) { best = md; bi = n; }
    }
    rv[tid] = best; ri[tid] = bi;
    __syncthreads();
    for (int s = 128; s > 0; s >>= 1) {
      if (tid < s) {
        float ov = rv[tid+s]; int oi = ri[tid+s];
        if (ov > rv[tid] || (ov == rv[tid] && oi < ri[tid])) { rv[tid] = ov; ri[tid] = oi; }
      }
      __syncthreads();
    }
    last = ri[0];
    __syncthreads();
  }
}

// ---------------- ball query: one wave32 per center, ordered first-K ----------------
__global__ __launch_bounds__(256) void ballq_kernel(const float* __restrict__ p,
                                                    const float* __restrict__ centers,
                                                    int* __restrict__ gidx) {
  int wave = threadIdx.x >> 5, lane = threadIdx.x & 31;
  int cidx = blockIdx.x * 8 + wave;
  int b = cidx / MCENT;
  const float* pb = p + (size_t)b * NN * 3;
  float cx = centers[cidx*3+0], cy = centers[cidx*3+1], cz = centers[cidx*3+2];
  int found = 0, firstIdx = 0;
  bool haveFirst = false;
  for (int n0 = 0; n0 < NN; n0 += 32) {
    int n = n0 + lane;
    float dx = pb[n*3+0]-cx, dy = pb[n*3+1]-cy, dz = pb[n*3+2]-cz;
    bool in = (dx*dx + dy*dy + dz*dz) < R2;
    unsigned mask = (unsigned)__ballot(in);          // wave32: low 32 bits
    if (!haveFirst && mask) { firstIdx = n0 + __builtin_ctz(mask); haveFirst = true; }
    if (in) {
      int pos = found + __popc(mask & ((1u << lane) - 1u));
      if (pos < KNB) gidx[cidx*KNB + pos] = n;
    }
    found += __popc(mask);                           // uniform
    if (found >= KNB) break;                         // uniform exit
  }
  for (int t = found + lane; t < KNB; t += 32) gidx[cidx*KNB + t] = firstIdx;
}

// ---------------- build F0 [32][L] bf16 (6 channels + zero pad) ----------------
__global__ __launch_bounds__(256) void build_f0(const float* __restrict__ p,
                                                const float* __restrict__ x,
                                                const float* __restrict__ centers,
                                                const int* __restrict__ gidx,
                                                const int* __restrict__ cpt,
                                                __bf16* __restrict__ F0) {
  int l = blockIdx.x * 256 + threadIdx.x;   // column = (b*M+m)*K + k
  int cidx = l >> 5;
  int b = cidx >> 9;                        // M = 512
  int g = gidx[l];
  int ci = cpt[cidx];
  const float* pb = p + (size_t)b * NN * 3;
  const float* xb = x + (size_t)b * 3 * NN;
  #pragma unroll
  for (int c = 0; c < 3; ++c) {
    F0[(size_t)c * LCOLS + l]     = (__bf16)(pb[g*3 + c] - centers[cidx*3 + c]);
    F0[(size_t)(3+c) * LCOLS + l] = (__bf16)(xb[c*NN + g] - xb[c*NN + ci]);
  }
  #pragma unroll
  for (int c = 6; c < 32; ++c) F0[(size_t)c * LCOLS + l] = (__bf16)0.0f;
}

// ---------------- WMMA GEMM: Y[O][L] = W[O][C] @ X[C][L] ----------------
// block = 8 waves; block tile = 128 rows x 64 cols; per-wave 16 rows x 64 cols
// (4 WMMAs per 32-deep K-chunk, A fragment reused 4x).
// W tile staged via async global->LDS b128 (ASYNCcnt) when available;
// X tile staged transposed ([64 cols][32 c] bf16) so fragments are ds_load_b128.
__global__ __launch_bounds__(256) void gemm_wmma_bf16(const float* __restrict__ W,
                                                      const __bf16* __restrict__ X,
                                                      float* __restrict__ Y,
                                                      int C) {
  __shared__ __align__(16) float  Wl[128 * 32];   // 16 KB weight tile (f32)
  __shared__ __align__(16) __bf16 XT[64 * 32];    // 4 KB transposed activation tile
  int tid  = threadIdx.x;
  int wave = tid >> 5, lane = tid & 31;
  int g = lane >> 4, ln = lane & 15;              // ln = A row (m) and B col (n)
  int oBase   = blockIdx.y * 128;
  int colBase = blockIdx.x * 64;

  // staging maps
  int wr  = tid >> 1,        wc0 = (tid & 1) * 16;   // W: row, 16-float slab
  int xc  = tid >> 3,        xj8 = (tid & 7) * 8;    // X: c-row, 8-col slab

  v8f acc0 = {}, acc1 = {}, acc2 = {}, acc3 = {};

  for (int c0 = 0; c0 < C; c0 += 32) {
    __syncthreads();   // previous iteration's fragment reads complete

    // ---- stage W[oBase..+127][c0..+31] into LDS ----
    {
      const float* gsrc = W + (size_t)(oBase + wr) * C + (c0 + wc0);
      float*       ldst = &Wl[wr * 32 + wc0];
#if USE_ASYNC
      #pragma unroll
      for (int q = 0; q < 4; ++q)
        __builtin_amdgcn_global_load_async_to_lds_b128(
            (gv4i_p)(gsrc + q * 4),
            (sv4i_p)(ldst + q * 4), 0, 0);
#else
      #pragma unroll
      for (int q = 0; q < 16; ++q) ldst[q] = gsrc[q];
#endif
    }
    // ---- stage X[c0..+31][colBase..+63] transposed into LDS ----
    {
      v8bf v = *(const v8bf*)(X + (size_t)(c0 + xc) * LCOLS + colBase + xj8);
      #pragma unroll
      for (int j = 0; j < 8; ++j) XT[(xj8 + j) * 32 + xc] = v[j];
    }
#if USE_ASYNC
    __builtin_amdgcn_s_wait_asynccnt(0);
#endif
    __syncthreads();

    // ---- compose A fragment (contiguous K = 16g + e order, same as B frag) ----
    v16bf a;
    {
      const float* wrow = &Wl[(wave * 16 + ln) * 32];
      #pragma unroll
      for (int e = 0; e < 8; ++e) {
        a[e]     = (__bf16)wrow[g * 16 + e];         // K = e + g*16
        a[e + 8] = (__bf16)wrow[g * 16 + 8 + e];     // K = e + 8 + g*16
      }
    }
    // ---- 4 column subtiles: B frag is two contiguous b128 LDS loads ----
    #pragma unroll
    for (int t = 0; t < 4; ++t) {
      const __bf16* xcol = &XT[(t * 16 + ln) * 32 + g * 16];   // K = e + g*16
      v8bf blo = *(const v8bf*)(xcol);
      v8bf bhi = *(const v8bf*)(xcol + 8);
      v16bf bf;
      #pragma unroll
      for (int e = 0; e < 8; ++e) { bf[e] = blo[e]; bf[e + 8] = bhi[e]; }
      if (t == 0) acc0 = __builtin_amdgcn_wmma_f32_16x16x32_bf16(false, a, false, bf, (short)0, acc0, false, false);
      if (t == 1) acc1 = __builtin_amdgcn_wmma_f32_16x16x32_bf16(false, a, false, bf, (short)0, acc1, false, false);
      if (t == 2) acc2 = __builtin_amdgcn_wmma_f32_16x16x32_bf16(false, a, false, bf, (short)0, acc2, false, false);
      if (t == 3) acc3 = __builtin_amdgcn_wmma_f32_16x16x32_bf16(false, a, false, bf, (short)0, acc3, false, false);
    }
  }

  // ---- epilogue: C/D map M = r + g*8 ----
  #pragma unroll
  for (int r = 0; r < 8; ++r) {
    size_t row = (size_t)(oBase + wave * 16 + g * 8 + r) * LCOLS;
    Y[row + colBase +  0 + ln] = acc0[r];
    Y[row + colBase + 16 + ln] = acc1[r];
    Y[row + colBase + 32 + ln] = acc2[r];
    Y[row + colBase + 48 + ln] = acc3[r];
  }
}

// ---------------- per-channel mean / inv-std over L columns ----------------
__global__ __launch_bounds__(256) void chan_stats(const float* __restrict__ Y,
                                                  float* __restrict__ stats) {
  __shared__ float ssum[256], ssq[256];
  int ch = blockIdx.x, tid = threadIdx.x;
  const float* row = Y + (size_t)ch * LCOLS;
  float s = 0.f, q = 0.f;
  for (int i = tid; i < LCOLS; i += 256) { float v = row[i]; s += v; q += v*v; }
  ssum[tid] = s; ssq[tid] = q;
  __syncthreads();
  for (int st = 128; st > 0; st >>= 1) {
    if (tid < st) { ssum[tid] += ssum[tid+st]; ssq[tid] += ssq[tid+st]; }
    __syncthreads();
  }
  if (tid == 0) {
    float mean = ssum[0] / (float)LCOLS;
    float var  = ssq[0]  / (float)LCOLS - mean*mean;
    stats[ch]       = mean;
    stats[512 + ch] = rsqrtf(var + EPSV);
  }
}

// ---------------- fused BN + ReLU -> bf16 ----------------
__global__ __launch_bounds__(256) void bn_relu_bf16(const float* __restrict__ Y,
                                                    const float* __restrict__ stats,
                                                    const float* __restrict__ gamma,
                                                    const float* __restrict__ beta,
                                                    __bf16* __restrict__ H) {
  size_t i = (size_t)blockIdx.x * 256 + threadIdx.x;
  int o = (int)(i >> 17);                            // L = 2^17
  float v = (Y[i] - stats[o]) * stats[512 + o] * gamma[o] + beta[o];
  H[i] = (__bf16)fmaxf(v, 0.f);
}

// ---------------- K-max of (Y + bias) per (b,o,m) ----------------
__global__ __launch_bounds__(256) void kmax_bias(const float* __restrict__ Y,
                                                 const float* __restrict__ bias,
                                                 float* __restrict__ outv) {
  int id = blockIdx.x * 256 + threadIdx.x;           // 2^20 ids
  int m = id & 511, o = (id >> 9) & 255, b = id >> 17;
  size_t base = (size_t)o * LCOLS + (((size_t)(b*MCENT + m)) << 5);
  float bv = bias[o], v = -1e30f;
  #pragma unroll
  for (int k = 0; k < KNB; ++k) v = fmaxf(v, Y[base + k] + bv);
  outv[(((size_t)b << 8) + o) * MCENT + m] = v;
}

// ---------------- build F2 [512][L] = [fmax bcast ; Y1 + b1] in bf16 ----------------
__global__ __launch_bounds__(256) void build_f2(const float* __restrict__ Y1,
                                                const float* __restrict__ b1,
                                                const float* __restrict__ fmaxb,
                                                __bf16* __restrict__ F2) {
  size_t i = (size_t)blockIdx.x * 256 + threadIdx.x; // 512*L elements
  int o = (int)(i >> 17);
  int l = (int)(i & (LCOLS - 1));
  float v;
  if (o < 256) {
    int b = l >> 14, m = (l >> 5) & 511;             // M*K = 2^14
    v = fmaxb[(((size_t)b << 8) + o) * MCENT + m];
  } else {
    v = Y1[(size_t)(o - 256) * LCOLS + l] + b1[o - 256];
  }
  F2[i] = (__bf16)v;
}

// =====================================================================
extern "C" void kernel_launch(void* const* d_in, const int* in_sizes, int n_in,
                              void* d_out, int out_size, void* d_ws, size_t ws_size,
                              hipStream_t stream) {
  (void)in_sizes; (void)n_in; (void)out_size; (void)ws_size;
  const float* p   = (const float*)d_in[0];
  const float* x   = (const float*)d_in[1];
  const float* W0  = (const float*)d_in[2];
  const float* g0  = (const float*)d_in[3];
  const float* be0 = (const float*)d_in[4];
  const float* W1  = (const float*)d_in[5];
  const float* b1  = (const float*)d_in[6];
  const float* W2  = (const float*)d_in[7];
  const float* g2  = (const float*)d_in[8];
  const float* be2 = (const float*)d_in[9];
  const float* W3  = (const float*)d_in[10];
  const float* b3  = (const float*)d_in[11];
  float* out = (float*)d_out;

  float* out_p = out;                 // [B,N,3]   196608
  float* out_c = out + 196608;        // [B,M,3]   12288
  float* out_x = out + 208896;        // [B,3,N]   196608
  float* out_f = out + 405504;        // [B,256,M] 1048576

  char* ws = (char*)d_ws;
  float*  bufA  = (float*)ws;                                   // 512*L*4 = 256 MB (Y0/Y1/Y2/Y3)
  __bf16* bufB  = (__bf16*)(ws + (size_t)512 * LCOLS * 4);      // 512*L*2 = 128 MB (F0/H0/F2/H2)
  int*    gidx  = (int*)(ws + (size_t)512 * LCOLS * 6);         // L ints
  int*    cpt   = gidx + LCOLS;                                  // B*M ints
  float*  fmaxb = (float*)(cpt + BB * MCENT);                    // 2^20 floats
  float*  stats = fmaxb + (1 << 20);                             // 1024 floats
  float*  Wpad  = stats + 1024;                                  // 256*32 floats

  // pass-through outputs + sampling
  copy_f32<<<768, 256, 0, stream>>>(p, out_p, 196608);
  copy_f32<<<768, 256, 0, stream>>>(x, out_x, 196608);
  pad_w0<<<32, 256, 0, stream>>>(W0, Wpad);
  fps_kernel<<<BB, 256, 0, stream>>>(p, cpt, out_c);
  ballq_kernel<<<BB*MCENT/8, 256, 0, stream>>>(p, out_c, gidx);
  build_f0<<<LCOLS/256, 256, 0, stream>>>(p, x, out_c, gidx, cpt, bufB);

  // conv1: W0 (padded to C=32) -> BN+ReLU -> W1
  gemm_wmma_bf16<<<dim3(LCOLS/64, 2), 256, 0, stream>>>(Wpad, bufB, bufA, 32);
  chan_stats<<<256, 256, 0, stream>>>(bufA, stats);
  bn_relu_bf16<<<256*(LCOLS/256), 256, 0, stream>>>(bufA, stats, g0, be0, bufB);
  gemm_wmma_bf16<<<dim3(LCOLS/64, 2), 256, 0, stream>>>(W1, bufB, bufA, 256);

  // mid max-pool concat
  kmax_bias<<<(1<<20)/256, 256, 0, stream>>>(bufA, b1, fmaxb);
  build_f2<<<512*(LCOLS/256), 256, 0, stream>>>(bufA, b1, fmaxb, bufB);

  // conv2: W2 -> BN+ReLU -> W3 -> final K-max
  gemm_wmma_bf16<<<dim3(LCOLS/64, 4), 256, 0, stream>>>(W2, bufB, bufA, 512);
  chan_stats<<<512, 256, 0, stream>>>(bufA, stats);
  bn_relu_bf16<<<512*(LCOLS/256), 256, 0, stream>>>(bufA, stats, g2, be2, bufB);
  gemm_wmma_bf16<<<dim3(LCOLS/64, 2), 256, 0, stream>>>(W3, bufB, bufA, 512);
  kmax_bias<<<(1<<20)/256, 256, 0, stream>>>(bufA, b3, out_f);
}
